// SwitchHeadCore_39694087749774
// MI455X (gfx1250) — compile-verified, matrix operand order checked by reading, not run on hardware
//
#include <hip/hip_runtime.h>

typedef __bf16 bf16;
typedef __attribute__((ext_vector_type(16))) __bf16 v16bf;
typedef __attribute__((ext_vector_type(8)))  __bf16 v8bf;
typedef __attribute__((ext_vector_type(8)))  float   v8f;
typedef __attribute__((ext_vector_type(4))) unsigned int u32x4;
typedef __attribute__((ext_vector_type(4))) int          i32x4;
typedef __attribute__((ext_vector_type(8))) int          i32x8;

#if defined(__has_builtin)
#if __has_builtin(__builtin_amdgcn_tensor_load_to_lds)
#define HAVE_TDM 1
#endif
#endif
#ifndef HAVE_TDM
#define HAVE_TDM 0
#endif

constexpr int kB = 4, kS = 1024, kD = 1024, kH = 8, kE = 4, kDH = 128;
constexpr int kHE   = kH * kE;     // 32
constexpr int kHDH  = kH * kDH;    // 1024
constexpr int kHEDH = kHE * kDH;   // 4096
constexpr int kTOK  = kB * kS;     // 4096

// ---------------------------------------------------------------- helpers
__device__ inline v8f wmma_bf16(v16bf a, v16bf b, v8f c) {
  // D = A(16x32 bf16) * B(32x16 bf16) + C(16x16 f32)
  return __builtin_amdgcn_wmma_f32_16x16x32_bf16(
      false, a, false, b, (short)0, c, false, false);
}

__device__ inline v16bf frag2(const bf16* p0, const bf16* p1) {
  union { v16bf v; v8bf h[2]; } u;
  u.h[0] = *(const v8bf*)p0;   // 16B ds/global load
  u.h[1] = *(const v8bf*)p1;
  return u.v;
}

__device__ inline void stc(float* p, float v) { *p = v; }
__device__ inline void stc(bf16*  p, float v) { *p = (bf16)v; }

// ---------------------------------------------------------------- prep
__global__ void f32_to_bf16(const float* __restrict__ in, bf16* __restrict__ out, int n) {
  int i = blockIdx.x * blockDim.x + threadIdx.x;
  if (i < n) out[i] = (bf16)in[i];
}

// Wv [HE, D, DH] -> Wvt [(he*DH + c), d]  (B^T for the V-expert GEMM)
__global__ void transpose_wv_k(const float* __restrict__ Wv, bf16* __restrict__ Wvt) {
  long idx = (long)blockIdx.x * blockDim.x + threadIdx.x;   // HE*DH*D elems
  int  d   = (int)(idx & (kD - 1));
  long nr  = idx >> 10;          // he*DH + c
  int  c   = (int)(nr & (kDH - 1));
  long he  = nr >> 7;
  Wvt[idx] = (bf16)Wv[(he * kD + d) * kDH + c];
}

// Wo [HE, DH, D] -> Wot [d, (he*DH + c)]  (B^T for the output GEMM)
__global__ void transpose_wo_k(const float* __restrict__ Wo, bf16* __restrict__ Wot) {
  long idx = (long)blockIdx.x * blockDim.x + threadIdx.x;   // D*HE*DH elems
  int  hec = (int)(idx & (kHEDH - 1));
  long d   = idx >> 12;
  Wot[idx] = (bf16)Wo[(long)hec * kD + d];
}

// ---------------------------------------------------------------- routing
// one wave per token; lane = h*E+e; sigmoid gate, keep top-2 of each 4-group
__global__ void route_topk(const float* __restrict__ src, const float* __restrict__ sel,
                           float* __restrict__ w) {
  const int token = blockIdx.x * 8 + (threadIdx.x >> 5);
  const int lane  = threadIdx.x & 31;
  const float* s  = src + (long)token * kD;
  const float* sl = sel + (long)lane * kD;
  float acc = 0.f;
  for (int d = 0; d < kD; d += 4)
    acc += s[d]*sl[d] + s[d+1]*sl[d+1] + s[d+2]*sl[d+2] + s[d+3]*sl[d+3];
  float g = 1.f / (1.f + __expf(-acc));
  const int e = lane & 3, base = lane & ~3;
  int rank = 0;
#pragma unroll
  for (int j = 0; j < 4; ++j) {
    float gj = __shfl(g, base + j, 32);
    if (j != e && (gj > g || (gj == g && j < e))) ++rank;
  }
  w[(long)token * kHE + lane] = (rank < 2) ? g : 0.f;
}

// ---------------------------------------------------------------- GEMM
// C[M,N] (+scale) = A[M,K]bf16 * Bt[N,K]bf16^T ; 64x64 block tile, 8 waves,
// software-pipelined global->LDS staging + prefetch of the k+2 slice.
template <typename OutT>
__global__ void gemm_bf16_wmma(const bf16* __restrict__ A, const bf16* __restrict__ Bt,
                               OutT* __restrict__ C, int M, int N, int Kd, float scale) {
  __shared__ bf16 As[64][40];   // 80B rows keep 16B chunks aligned
  __shared__ bf16 Bs[64][40];
  const int tid  = threadIdx.x;
  const int wave = tid >> 5, lane = tid & 31;
  const int p = lane & 15, hi = lane >> 4;
  const int tn = wave & 3, tm = wave >> 2;           // tiles (tm,tn) and (tm+2,tn)
  const long bm = (long)blockIdx.y * 64, bn = (long)blockIdx.x * 64;
  const int lrow = tid >> 2, lcol = (tid & 3) * 8;
  const v8f z = {0,0,0,0,0,0,0,0};
  v8f acc0 = z, acc1 = z;

  const bf16* pa = &A [(bm + lrow) * (long)Kd + lcol];
  const bf16* pb = &Bt[(bn + lrow) * (long)Kd + lcol];
  v8bf aReg = *(const v8bf*)pa;
  v8bf bReg = *(const v8bf*)pb;

  for (int k0 = 0; k0 < Kd; k0 += 32) {
    *(v8bf*)&As[lrow][lcol] = aReg;
    *(v8bf*)&Bs[lrow][lcol] = bReg;
    __syncthreads();
    if (k0 + 32 < Kd) {                 // fetch next slice during the WMMAs
      aReg = *(const v8bf*)(pa + k0 + 32);
      bReg = *(const v8bf*)(pb + k0 + 32);
      if (k0 + 64 < Kd) {               // and prefetch the one after
        __builtin_prefetch(pa + k0 + 64, 0, 1);
        __builtin_prefetch(pb + k0 + 64, 0, 1);
      }
    }
    v16bf bq = frag2(&Bs[tn*16 + p][hi*8],       &Bs[tn*16 + p][16 + hi*8]);
    v16bf a0 = frag2(&As[tm*16 + p][hi*8],       &As[tm*16 + p][16 + hi*8]);
    v16bf a1 = frag2(&As[(tm+2)*16 + p][hi*8],   &As[(tm+2)*16 + p][16 + hi*8]);
    acc0 = wmma_bf16(a0, bq, acc0);
    acc1 = wmma_bf16(a1, bq, acc1);
    __syncthreads();
  }
#pragma unroll
  for (int r = 0; r < 8; ++r) {
    long col = bn + tn*16 + p;
    stc(&C[(bm + tm*16     + hi*8 + r) * (long)N + col], acc0[r] * scale);
    stc(&C[(bm + (tm+2)*16 + hi*8 + r) * (long)N + col], acc1[r] * scale);
  }
}

// ---------------------------------------------------------------- V mixture
__global__ void weight_v(const bf16* __restrict__ v_all, const float* __restrict__ wv,
                         bf16* __restrict__ vout) {
  long idx = (long)blockIdx.x * blockDim.x + threadIdx.x;  // B*H*S*DH
  int c = (int)(idx & (kDH - 1));
  int s = (int)((idx >> 7)  & (kS - 1));
  int h = (int)((idx >> 17) & (kH - 1));
  int b = (int)( idx >> 20);
  long token = (long)b * kS + s;
  float acc = 0.f;
#pragma unroll
  for (int e = 0; e < kE; ++e) {
    float g = wv[(token * kH + h) * kE + e];
    acc += g * (float)v_all[token * kHEDH + (h*kE + e)*kDH + c];
  }
  vout[idx] = (bf16)acc;
}

// ---------------------------------------------------------------- attention
// flash-style: block = (qtile of 64 rows, h, b); 4 waves x 16 query rows.
// K tile is DMA'd by the Tensor Data Mover (with LDS padding so it matches the
// padded Ks layout); V tile is loaded+transposed manually for the P*V WMMA.
__global__ void __launch_bounds__(128, 1)
attention_fa(const bf16* __restrict__ qb, const bf16* __restrict__ kb,
             const bf16* __restrict__ vb, bf16* __restrict__ res) {
  __shared__ bf16 Ks[32][136];     // 32 keys x 128 ch (+4dw pad per 64dw row)
  __shared__ bf16 Vt[128][40];     // transposed V tile: [c][key]
  __shared__ bf16 Ps[4][16][40];   // per-wave P relayout buffer
  const int tid  = threadIdx.x;    // 0..127
  const int wave = tid >> 5, lane = tid & 31;
  const int p = lane & 15, hi = lane >> 4;
  const int qt = blockIdx.x, h = blockIdx.y, b = blockIdx.z;

  // Q A-fragments, kept in registers (K-dim 128 = 4 frags of 32)
  const long qrow = (long)b * kS + qt*64 + wave*16 + p;
  v16bf qf[4];
#pragma unroll
  for (int ck = 0; ck < 4; ++ck)
    qf[ck] = frag2(&qb[qrow*kHDH + h*kDH + ck*32 + hi*8],
                   &qb[qrow*kHDH + h*kDH + ck*32 + 16 + hi*8]);

  const v8f z = {0,0,0,0,0,0,0,0};
  v8f O[8];
  float m[8], l[8];
#pragma unroll
  for (int g = 0; g < 8; ++g) O[g] = z;
#pragma unroll
  for (int r = 0; r < 8; ++r) { m[r] = -3.0e38f; l[r] = 0.f; }

  for (int kt = 0; kt < kS/32; ++kt) {
#if HAVE_TDM
    // --- K tile: TDM 2D descriptor, 32 rows x 256B, row stride H*DH*2 bytes,
    //     pad_enable: +4 DWORDs after every 64 DWORDs -> Ks[32][136] layout.
    if (wave == 0) {
      unsigned lds_off = (unsigned)(size_t)&Ks[0][0];
      unsigned long long ga =
          (unsigned long long)(size_t)&kb[((long)b*kS + (long)kt*32)*kHDH + h*kDH];
      u32x4 g0 = {1u,                                  // count=1 (valid D#)
                  lds_off,                             // lds_addr
                  (unsigned)ga,                        // global_addr[31:0]
                  (unsigned)(ga >> 32) | 0x80000000u}; // global_addr hi | type=2
      i32x8 g1 = {(int)0x07510000u,   // data_size=2B, pad_en, intv=64dw, amt=4dw
                  (int)(128u << 16),  // tensor_dim0 = DH (bits 79:48)
                  (int)(1024u << 16), // tensor_dim1 = S  (bits 111:80)
                  (int)(128u << 16),  // tile_dim0 = DH   (bits 127:112)
                  32,                 // tile_dim1 = 32 keys
                  1024,               // tensor_dim0_stride = H*DH elems
                  0, 0};
      i32x4 g2 = {0, 0, 0, 0};
      i32x4 g3 = {0, 0, 0, 0};
#if defined(__clang_major__) && __clang_major__ >= 23
      i32x8 g4 = {0, 0, 0, 0, 0, 0, 0, 0};
      __builtin_amdgcn_tensor_load_to_lds(g0, g1, g2, g3, g4, 0);
#else
      __builtin_amdgcn_tensor_load_to_lds(g0, g1, g2, g3, 0);
#endif
      __builtin_amdgcn_s_wait_tensorcnt(0);
    }
#endif
#pragma unroll
    for (int i = 0; i < 4; ++i) {
      int chunk = tid * 4 + i;            // 512 chunks of 8 bf16
      int row = chunk >> 4;               // key 0..31
      int c8  = (chunk & 15) * 8;
      long key = (long)kt*32 + row;
#if !HAVE_TDM
      *(v8bf*)&Ks[row][c8] =
          *(const v8bf*)&kb[((long)b*kS + key)*kHDH + h*kDH + c8];
#endif
      v8bf vv = *(const v8bf*)&vb[(((long)b*kH + h)*kS + key)*kDH + c8];
#pragma unroll
      for (int j = 0; j < 8; ++j) Vt[c8 + j][row] = vv[j];
    }
    if (kt + 1 < kS/32) {                 // prefetch next tile
      int chunk0 = tid * 4;
      int nrow = chunk0 >> 4, nc8 = (chunk0 & 15) * 8;
      __builtin_prefetch(&vb[(((long)b*kH + h)*kS + (kt+1)*32 + nrow)*kDH + nc8], 0, 1);
#if !HAVE_TDM
      __builtin_prefetch(&kb[((long)b*kS + (kt+1)*32 + nrow)*kHDH + h*kDH + nc8], 0, 1);
#endif
    }
    __syncthreads();

    // S = Q * K^T : two 16x16 score tiles (keys 0-15, 16-31)
    v8f sc[2];
#pragma unroll
    for (int jn = 0; jn < 2; ++jn) {
      v8f acc = z;
#pragma unroll
      for (int ck = 0; ck < 4; ++ck) {
        v16bf kf = frag2(&Ks[jn*16 + p][ck*32 + hi*8],
                         &Ks[jn*16 + p][ck*32 + 16 + hi*8]);
        acc = wmma_bf16(qf[ck], kf, acc);
      }
      sc[jn] = acc;
    }

    // online softmax (row = accum row of this lane half)
#pragma unroll
    for (int r = 0; r < 8; ++r) {
      float mx = fmaxf(sc[0][r], sc[1][r]);
#pragma unroll
      for (int d = 1; d < 16; d <<= 1) mx = fmaxf(mx, __shfl_xor(mx, d, 32));
      float mn    = fmaxf(m[r], mx);
      float alpha = __expf(m[r] - mn);
      float p0 = __expf(sc[0][r] - mn);
      float p1 = __expf(sc[1][r] - mn);
      float rs = p0 + p1;
#pragma unroll
      for (int d = 1; d < 16; d <<= 1) rs += __shfl_xor(rs, d, 32);
      l[r] = l[r] * alpha + rs;
      m[r] = mn;
#pragma unroll
      for (int g = 0; g < 8; ++g) O[g][r] *= alpha;
      int prow = hi*8 + r;                 // C-layout -> row-major P in LDS
      Ps[wave][prow][p]      = (bf16)p0;
      Ps[wave][prow][16 + p] = (bf16)p1;
    }

    // O += P(16x32) * V(32x128) ; same-wave LDS store->load is in-order
    v16bf pf = frag2(&Ps[wave][p][hi*8], &Ps[wave][p][16 + hi*8]);
#pragma unroll
    for (int g = 0; g < 8; ++g) {
      v16bf vf = frag2(&Vt[g*16 + p][hi*8], &Vt[g*16 + p][16 + hi*8]);
      O[g] = wmma_bf16(pf, vf, O[g]);
    }
    __syncthreads();
  }

  // store res [B,S,H,DH] bf16
#pragma unroll
  for (int g = 0; g < 8; ++g) {
#pragma unroll
    for (int r = 0; r < 8; ++r) {
      int srow = qt*64 + wave*16 + hi*8 + r;
      int c    = g*16 + p;
      float ov = O[g][r] / l[r];
      res[(((long)b*kS + srow)*kH + h)*kDH + c] = (bf16)ov;
    }
  }
}

// ---------------------------------------------------------------- O gating
__global__ void build_resw(const bf16* __restrict__ res, const float* __restrict__ wo,
                           bf16* __restrict__ resw) {
  long idx = (long)blockIdx.x * blockDim.x + threadIdx.x;  // TOK*HE*DH
  int  c  = (int)(idx & (kDH - 1));
  int  he = (int)((idx >> 7) & (kHE - 1));
  long token = idx >> 12;
  int h = he >> 2, e = he & 3;
  float g = wo[(token * kH + h) * kE + e];
  float r = (float)res[token * kHDH + h * kDH + c];
  resw[idx] = (bf16)(r * g);
}

// ---------------------------------------------------------------- launcher
extern "C" void kernel_launch(void* const* d_in, const int* in_sizes, int n_in,
                              void* d_out, int out_size, void* d_ws, size_t ws_size,
                              hipStream_t stream) {
  (void)in_sizes; (void)n_in; (void)out_size; (void)ws_size;
  const float* q_src = (const float*)d_in[0];
  const float* k_src = (const float*)d_in[1];
  const float* v_src = (const float*)d_in[2];
  const float* Wq    = (const float*)d_in[3];
  const float* Wk    = (const float*)d_in[4];
  const float* Wv    = (const float*)d_in[5];
  const float* Wo    = (const float*)d_in[6];
  const float* sel_v = (const float*)d_in[7];
  const float* sel_o = (const float*)d_in[8];

  char* base = (char*)d_ws;
  size_t off = 0;
  auto alloc = [&](size_t nbytes) -> void* {
    void* r = base + off;
    off = (off + nbytes + 255) & ~(size_t)255;
    return r;
  };
  bf16*  qs   = (bf16*) alloc((size_t)kTOK  * kD    * 2);
  bf16*  ksb  = (bf16*) alloc((size_t)kTOK  * kD    * 2);
  bf16*  vs   = (bf16*) alloc((size_t)kTOK  * kD    * 2);
  bf16*  Wqb  = (bf16*) alloc((size_t)kHDH  * kD    * 2);
  bf16*  Wkb  = (bf16*) alloc((size_t)kHDH  * kD    * 2);
  bf16*  Wvt  = (bf16*) alloc((size_t)kHEDH * kD    * 2);
  bf16*  Wot  = (bf16*) alloc((size_t)kD    * kHEDH * 2);
  bf16*  qpb  = (bf16*) alloc((size_t)kTOK  * kHDH  * 2);
  bf16*  kpb  = (bf16*) alloc((size_t)kTOK  * kHDH  * 2);
  bf16*  vall = (bf16*) alloc((size_t)kTOK  * kHEDH * 2);  // reused as resw
  bf16*  vbuf = (bf16*) alloc((size_t)kB*kH*kS*kDH  * 2);
  bf16*  resb = (bf16*) alloc((size_t)kTOK  * kHDH  * 2);
  float* wv   = (float*)alloc((size_t)kTOK  * kHE   * 4);
  float* wo   = (float*)alloc((size_t)kTOK  * kHE   * 4);
  bf16*  resw = vall;   // lifetimes disjoint: vall dead after weight_v

  const int n4M = kTOK * kD;                 // 4,194,304
  f32_to_bf16<<<n4M/256, 256, 0, stream>>>(q_src, qs,  n4M);
  f32_to_bf16<<<n4M/256, 256, 0, stream>>>(k_src, ksb, n4M);
  f32_to_bf16<<<n4M/256, 256, 0, stream>>>(v_src, vs,  n4M);
  const int nW = kHDH * kD;                  // 1,048,576
  f32_to_bf16<<<nW/256, 256, 0, stream>>>(Wq, Wqb, nW);
  f32_to_bf16<<<nW/256, 256, 0, stream>>>(Wk, Wkb, nW);
  transpose_wv_k<<<n4M/256, 256, 0, stream>>>(Wv, Wvt);
  transpose_wo_k<<<n4M/256, 256, 0, stream>>>(Wo, Wot);

  route_topk<<<kTOK/8, 256, 0, stream>>>(k_src, sel_v, wv);
  route_topk<<<kTOK/8, 256, 0, stream>>>(q_src, sel_o, wo);

  const float qkscale = 0.29730177875068026f;  // (1/sqrt(DH))^0.5
  gemm_bf16_wmma<bf16><<<dim3(kHDH/64,  kTOK/64), 256, 0, stream>>>(qs,  Wqb, qpb,  kTOK, kHDH,  kD, qkscale);
  gemm_bf16_wmma<bf16><<<dim3(kHDH/64,  kTOK/64), 256, 0, stream>>>(ksb, Wkb, kpb,  kTOK, kHDH,  kD, qkscale);
  gemm_bf16_wmma<bf16><<<dim3(kHEDH/64, kTOK/64), 256, 0, stream>>>(vs,  Wvt, vall, kTOK, kHEDH, kD, 1.f);

  weight_v<<<(kB*kH*kS*kDH)/256, 256, 0, stream>>>(vall, wv, vbuf);

  attention_fa<<<dim3(kS/64, kH, kB), 128, 0, stream>>>(qpb, kpb, vbuf, resb);

  build_resw<<<(long)(kTOK)*kHEDH/256, 256, 0, stream>>>(resb, wo, resw);

  gemm_bf16_wmma<float><<<dim3(kD/64, kTOK/64), 256, 0, stream>>>(resw, Wot, (float*)d_out, kTOK, kD, kHEDH, 1.f);
}
